// STBlock_62577673503436
// MI455X (gfx1250) — matrix operand order repaired and checked by reading, not compiled, not run on hardware
//
#include <hip/hip_runtime.h>
#include <hip/hip_bf16.h>

typedef __attribute__((ext_vector_type(16))) _Float16 v16h;
typedef __attribute__((ext_vector_type(8)))  _Float16 v8h;
typedef __attribute__((ext_vector_type(8)))  float    v8f;
typedef __attribute__((ext_vector_type(4)))  int      i32x4;
typedef __attribute__((ext_vector_type(8)))  int      i32x8;

#define BB 8
#define SS 64
#define NN 256
#define FF 16
#define HH 16
#define CC 4096      // NN*HH
#define SP 66        // S padded by 1 on each side for conv halo
#define ALPHA 0.2f
#define EPSV 1e-5f
#define NEGV -9000000000000000.0f

// ---------------------------------------------------------------------------
// Kernel 0: zero the conv halo columns (s = 0 and s = SP-1) of tT (f16)
// ---------------------------------------------------------------------------
__global__ void pad_kernel(_Float16* __restrict__ tT) {
  int t = blockIdx.x * blockDim.x + threadIdx.x;   // 0 .. BB*2*CC-1
  if (t < BB * 2 * CC) {
    int b   = t / (2 * CC);
    int r   = t % (2 * CC);
    int col = (r < CC) ? 0 : (SP - 1);
    int c   = r % CC;
    tT[((size_t)b * SP + col) * CC + c] = (_Float16)0.0f;
  }
}

// ---------------------------------------------------------------------------
// Kernel 1: convert conv_w f32 -> f16, pre-swizzled into WMMA B-fragment
// layout.  Tile = (kk, ciT, coT) covering K=32 (ci), N=16 (co).
// Within a tile: lane l, half p  ->  K = 16*(l/16)+p , N = l%16.
// Storage: wsB[tile*512 + lane*16 + p]  (each lane's 16 halfs contiguous).
// ---------------------------------------------------------------------------
__global__ void wconv_kernel(const float* __restrict__ conv_w,
                             _Float16* __restrict__ wsB) {
  int t    = blockIdx.x * 256 + threadIdx.x;
  int lane = t & 31;
  int tile = t >> 5;                       // 3 * (CC/32) * (CC/16) tiles
  int coT  = tile % (CC / 16);
  int rest = tile / (CC / 16);
  int ciT  = rest % (CC / 32);
  int kk   = rest / (CC / 32);
  int co      = coT * 16 + (lane & 15);
  int ciBase  = ciT * 32 + (lane >> 4) * 16;
  const float* src = conv_w + ((size_t)co * CC + ciBase) * 3 + kk;
  v16h v;
  #pragma unroll
  for (int p = 0; p < 16; ++p) v[p] = (_Float16)src[p * 3];
  *(v16h*)(wsB + (size_t)tile * 512 + lane * 16) = v;
}

// ---------------------------------------------------------------------------
// Kernel 2: GAT per (b,s).  8 waves x 32 lanes.
//   Phase A: Wh = x_bs(256x16) @ W(16x16) via wmma f16 (K padded 16->32)
//   Phase B: f1/f2 per node
//   Phase C: per-row online softmax stats (max, 1/sum)
//   Phase D: spatial = softmax(e) @ Wh via wmma f16, attn fragments built
//            on the fly as exp(e - rowmax); 1/rowsum folded into epilogue.
//   Output stored f16 transposed+padded: tT[b][s+1][node*16+h]
// ---------------------------------------------------------------------------
__global__ void __launch_bounds__(256) gat_kernel(
    const float* __restrict__ x, const float* __restrict__ adj,
    const float* __restrict__ W, const float* __restrict__ a1,
    const float* __restrict__ a2, _Float16* __restrict__ tT) {
  __shared__ float whLds[NN * HH];   // 16 KB
  __shared__ float f1s[NN], f2s[NN], ms[NN], invs[NN];

  int bs = blockIdx.x;               // (b,s)
  int b = bs / SS, s = bs % SS;
  const float* xbs = x + (size_t)bs * (NN * FF);
  int tid   = threadIdx.x;
  int wave  = tid >> 5;
  int lane  = tid & 31;
  int lhalf = lane >> 4;             // 0/1
  int lmod  = lane & 15;

  // ---- Phase A: Wh via WMMA (one 16x16x32 per 16-row tile, K zero-padded)
  for (int tt = 0; tt < 2; ++tt) {
    int i0  = (wave * 2 + tt) * 16;
    int row = i0 + lmod;
    v16h af = {};
    #pragma unroll
    for (int q = 0; q < 8; ++q)            // K = 8*lhalf + q  (< 16 valid)
      af[q] = (_Float16)xbs[row * FF + 8 * lhalf + q];
    v16h bf = {};
    {                                      // p>=8 covers K>=16 -> zero
      #pragma unroll
      for (int p = 0; p < 8; ++p)          // K = 8*lhalf + p, N = lmod
        bf[p] = (_Float16)W[(8 * lhalf + p) * HH + lmod];
    }
    v8f acc = {};
    acc = __builtin_amdgcn_wmma_f32_16x16x32_f16(false, af, false, bf,
                                                 (short)0, acc, false, false);
    #pragma unroll
    for (int r = 0; r < 8; ++r)
      whLds[(i0 + r + 8 * lhalf) * HH + lmod] = acc[r];
  }
  __syncthreads();

  // ---- Phase B: f1/f2 (one node per thread)
  {
    int i = tid;
    float s1 = 0.f, s2 = 0.f;
    #pragma unroll
    for (int h = 0; h < HH; ++h) {
      float w = whLds[i * HH + h];
      s1 += w * a1[h];
      s2 += w * a2[h];
    }
    f1s[i] = s1;
    f2s[i] = s2;
  }
  __syncthreads();

  // ---- Phase C: per-row online softmax stats
  {
    int i = tid;
    const float* arow = adj + (size_t)i * NN;
    float fi = f1s[i];
    float m = -3.0e38f, sum = 0.f;
    for (int j = 0; j < NN; ++j) {
      float e = fi + f2s[j];
      e = (e > 0.f) ? e : ALPHA * e;
      e = (arow[j] > 0.f) ? e : NEGV;
      float mn = fmaxf(m, e);
      sum = sum * __expf(m - mn) + __expf(e - mn);
      m = mn;
    }
    ms[i]   = m;
    invs[i] = 1.0f / sum;
  }
  __syncthreads();

  // ---- Phase D: spatial = attn @ Wh via WMMA, attn built on the fly
  for (int tt = 0; tt < 2; ++tt) {
    int i0  = (wave * 2 + tt) * 16;
    int row = i0 + lmod;
    float fi   = f1s[row];
    float mrow = ms[row];
    const float* arow = adj + (size_t)row * NN;
    v8f acc = {};
    for (int j0 = 0; j0 < NN; j0 += 32) {
      v16h af, bf;
      #pragma unroll
      for (int p = 0; p < 16; ++p) {
        int k = (p < 8) ? (j0 + 8 * lhalf + p)
                        : (j0 + 16 + 8 * lhalf + (p - 8));
        float e = fi + f2s[k];
        e = (e > 0.f) ? e : ALPHA * e;
        e = (arow[k] > 0.f) ? e : NEGV;
        af[p] = (_Float16)__expf(e - mrow);
        int kb = k;                        // B frag: same K as A
        bf[p] = (_Float16)whLds[kb * HH + lmod];
      }
      acc = __builtin_amdgcn_wmma_f32_16x16x32_f16(false, af, false, bf,
                                                   (short)0, acc, false, false);
    }
    // epilogue: scale by 1/rowsum, store f16 transposed+padded
    _Float16* dst = tT + ((size_t)b * SP + (s + 1)) * CC;
    #pragma unroll
    for (int r = 0; r < 8; ++r) {
      int node = i0 + r + 8 * lhalf;
      dst[node * HH + lmod] = (_Float16)(acc[r] * invs[node]);
    }
  }
}

// ---------------------------------------------------------------------------
// Kernel 3: conv1d as GEMM  O[s][co] = sum_{kk,ci} tT[b][s+kk][ci]*W[co][ci][kk]
// WG: 64(s) x 128(co) tile for one b.  Wave w -> co tile co0+16w, 4 s-tiles.
// A tile (64 s-rows x 64 ci) staged into LDS by the Tensor Data Mover
// (tensor_load_to_lds), double-buffered, overlapped with WMMA compute.
// Fused epilogue: +bias, BN, ReLU, +x residual, LayerNorm over h (=lane%16).
// ---------------------------------------------------------------------------
__global__ void __launch_bounds__(256) conv_kernel(
    const _Float16* __restrict__ tT, const _Float16* __restrict__ wsB,
    const float* __restrict__ x, const float* __restrict__ conv_b,
    const float* __restrict__ bn_gamma, const float* __restrict__ bn_beta,
    const float* __restrict__ bn_mean, const float* __restrict__ bn_var,
    const float* __restrict__ ln_gamma, const float* __restrict__ ln_beta,
    float* __restrict__ out) {
  __shared__ _Float16 aTile[2][64 * 64];   // 2 x 8 KB, double buffered

  int b    = blockIdx.y;
  int co0  = blockIdx.x * 128;
  int tid  = threadIdx.x;
  int wave = tid >> 5;
  int lane = tid & 31;
  int lhalf = lane >> 4, lmod = lane & 15;

  int coT  = (co0 >> 4) + wave;          // this wave's 16-wide co tile
  int myco = co0 + wave * 16 + lmod;

  // LDS byte offset of the staging buffers (low 32 bits of flat LDS addr)
  unsigned ldsBase = (unsigned)(uintptr_t)(&aTile[0][0]);

  // Tensor DMA descriptor group 1 (constant across stages):
  //   data_size=2B, tensor_dim0=CC, tensor_dim1=0x8000 (no clipping),
  //   tile_dim0=64 (ci, contiguous), tile_dim1=64 (s rows), stride0=CC.
  i32x8 g1;
  g1[0] = 0x00010000;            // workgroup_mask=0 | data_size=1 (2 bytes)
  g1[1] = (int)(CC << 16);       // tensor_dim0[15:0] << 16
  g1[2] = (int)0x80000000u;      // tensor_dim1[15:0]=0x8000 in high half
  g1[3] = (64 << 16);            // tensor_dim1[31:16]=0 | tile_dim0=64
  g1[4] = 64;                    // tile_dim1=64 | tile_dim2=0
  g1[5] = CC;                    // tensor_dim0_stride[31:0] (elements)
  g1[6] = 0;                     // stride0[47:32]=0 | stride1[15:0]=0
  g1[7] = 0;

  const int NST = 3 * (CC / 64); // 192 stages, 64 K-values each

  auto issue = [&](int stage) {
    int kk  = stage >> 6;
    int c64 = stage & 63;
    unsigned lds = ldsBase + (unsigned)((stage & 1) * (64 * 64 * 2));
    unsigned long long ga =
        (unsigned long long)(uintptr_t)tT +
        ((((unsigned long long)b) * SP + (unsigned)kk) * CC +
         (unsigned long long)c64 * 64) * 2ull;
    i32x4 g0;
    g0[0] = 1;                                        // count=1 (valid D#)
    g0[1] = (int)lds;                                 // lds_addr
    g0[2] = (int)(unsigned)ga;                        // global_addr[31:0]
    g0[3] = (int)((unsigned)(ga >> 32) | 0x80000000u);// ga[56:32] | type=2
    asm volatile("tensor_load_to_lds %0, %1" ::"s"(g0), "s"(g1) : "memory");
  };

  v8f acc[4] = {{}, {}, {}, {}};

  if (wave == 0) issue(0);
  for (int stage = 0; stage < NST; ++stage) {
    if (wave == 0) __builtin_amdgcn_s_wait_tensorcnt(0);
    __syncthreads();                     // stage data visible to all waves;
                                         // also: all waves done with buf^1
    if (wave == 0 && stage + 1 < NST) issue(stage + 1);

    const _Float16* atb = &aTile[stage & 1][0];
    int kk  = stage >> 6;
    int c64 = stage & 63;
    #pragma unroll
    for (int sub = 0; sub < 2; ++sub) {
      int ciT = c64 * 2 + sub;
      const _Float16* wp =
          wsB + ((size_t)(kk * (CC / 32) + ciT) * (CC / 16) + coT) * 512 +
          lane * 16;
      v16h bfrag = *(const v16h*)wp;                 // 32 B / lane (2x b128)
      __builtin_prefetch(wp + (size_t)2 * (CC / 16) * 512, 0, 1);
      #pragma unroll
      for (int tt = 0; tt < 4; ++tt) {
        int srow = tt * 16 + lmod;                   // A row = s
        const _Float16* ap = atb + srow * 64 + sub * 32 + 16 * lhalf;
        v8h alo = *(const v8h*)ap;                   // K = 16h   .. +8
        v8h ahi = *(const v8h*)(ap + 8);             // K = 16h+8 .. +8
        v16h afrag = __builtin_shufflevector(
            alo, ahi, 0, 1, 2, 3, 4, 5, 6, 7, 8, 9, 10, 11, 12, 13, 14, 15);
        acc[tt] = __builtin_amdgcn_wmma_f32_16x16x32_f16(
            false, afrag, false, bfrag, (short)0, acc[tt], false, false);
      }
    }
  }

  // ---- fused epilogue: bias + BN + ReLU + residual + LayerNorm(h)
  float cb = conv_b[myco];
  float bscale = bn_gamma[myco] * rsqrtf(bn_var[myco] + EPSV);
  float bmean = bn_mean[myco], bbeta = bn_beta[myco];
  float lg = ln_gamma[lmod], lb = ln_beta[lmod];

  #pragma unroll
  for (int tt = 0; tt < 4; ++tt) {
    #pragma unroll
    for (int r = 0; r < 8; ++r) {
      int srow = tt * 16 + r + 8 * lhalf;
      float y = acc[tt][r] + cb;
      y = (y - bmean) * bscale + bbeta;
      y = fmaxf(y, 0.f);
      size_t xi = ((size_t)(b * SS + srow)) * CC + myco;
      float v = y + x[xi];
      // LayerNorm over h = lmod: reduce across the 16-lane half-wave
      float sum = v, sq = v * v;
      #pragma unroll
      for (int m = 1; m < 16; m <<= 1) {
        sum += __shfl_xor(sum, m, 32);
        sq  += __shfl_xor(sq, m, 32);
      }
      float mu  = sum * (1.0f / 16.0f);
      float var = sq * (1.0f / 16.0f) - mu * mu;
      out[xi] = (v - mu) * rsqrtf(var + EPSV) * lg + lb;
    }
  }
}

// ---------------------------------------------------------------------------
extern "C" void kernel_launch(void* const* d_in, const int* in_sizes, int n_in,
                              void* d_out, int out_size, void* d_ws,
                              size_t ws_size, hipStream_t stream) {
  const float* x        = (const float*)d_in[0];
  const float* adj      = (const float*)d_in[1];
  const float* W        = (const float*)d_in[2];
  const float* a1       = (const float*)d_in[3];
  const float* a2       = (const float*)d_in[4];
  const float* conv_w   = (const float*)d_in[5];
  const float* conv_b   = (const float*)d_in[6];
  const float* bn_gamma = (const float*)d_in[7];
  const float* bn_beta  = (const float*)d_in[8];
  const float* bn_mean  = (const float*)d_in[9];
  const float* bn_var   = (const float*)d_in[10];
  const float* ln_gamma = (const float*)d_in[11];
  const float* ln_beta  = (const float*)d_in[12];
  float* out = (float*)d_out;

  // workspace: [tT f16: BB*SP*CC] [wsB f16: 3*CC*CC, 512B-aligned]
  _Float16* tT = (_Float16*)d_ws;
  size_t tT_elems = (size_t)BB * SP * CC;
  _Float16* wsB = tT + ((tT_elems + 255) & ~(size_t)255);

  pad_kernel<<<(BB * 2 * CC + 255) / 256, 256, 0, stream>>>(tT);
  wconv_kernel<<<(3 * (CC / 32) * (CC / 16) * 32) / 256, 256, 0, stream>>>(
      conv_w, wsB);
  gat_kernel<<<BB * SS, 256, 0, stream>>>(x, adj, W, a1, a2, tT);
  conv_kernel<<<dim3(CC / 128, BB), 256, 0, stream>>>(
      tT, wsB, x, conv_b, bn_gamma, bn_beta, bn_mean, bn_var, ln_gamma,
      ln_beta, out);
}
